// VectorQuantizer_17540646437246
// MI455X (gfx1250) — compile-verified, matrix operand order checked by reading, not run on hardware
//
#include <hip/hip_runtime.h>
#include <hip/hip_bf16.h>

// ---------------------------------------------------------------------------
// VQ-VAE vector quantizer, MI455X (gfx1250, wave32, WMMA + async LDS staging)
//   z_e:      [16, 2048, 512] f32   (d_in[0])
//   codebook: [4096, 512]     f32   (d_in[1])
// Outputs (flat, f32): z_q_st [N*D], k [N] (as float), vq_loss, utilization
// ---------------------------------------------------------------------------

typedef _Float16 half8  __attribute__((ext_vector_type(8)));
typedef _Float16 v16h   __attribute__((ext_vector_type(16)));
typedef _Float16 h4     __attribute__((ext_vector_type(4)));
typedef float    v8f    __attribute__((ext_vector_type(8)));
typedef float    f4     __attribute__((ext_vector_type(4)));

// type clang expects for the async-LDS builtin (from round-2 diagnostic):
// arg0: int __vector(4) __device__* ; arg1: LDS pointer ; then i32, i32
typedef int v4i __attribute__((vector_size(16)));
typedef __attribute__((address_space(1))) v4i* gas1_v4i;
typedef __attribute__((address_space(3))) v4i* las3_v4i;

#define NROWS   32768        // B*T
#define DDIM    512
#define KCODES  4096
#define ND      (NROWS * DDIM)
#define BETA    0.25f
#define NTILES  (KCODES / 16)   // 256
#define TILEH   (16 * DDIM)     // halfs per B tile (16 KB)

// workspace layout (byte offsets; ws is >=256B aligned from hipMalloc)
#define WS_LOSS   0                         // float
#define WS_USED   256                       // int[4096]
#define WS_WNORM  (256 + 16384)             // float[4096]
#define WS_CBH    (256 + 16384 + 16384)     // _Float16[4096*512]  (4 MB)

#if __has_builtin(__builtin_amdgcn_global_load_async_to_lds_b128) && \
    __has_builtin(__builtin_amdgcn_s_wait_asynccnt)
#define HAVE_ASYNC_LDS 1
#else
#define HAVE_ASYNC_LDS 0
#endif

// ---------------------------------------------------------------------------
__global__ void vq_init(float* loss, int* used) {
    int i = blockIdx.x * blockDim.x + threadIdx.x;
    if (i < KCODES) used[i] = 0;
    if (i == 0) *loss = 0.0f;
}

// one block per code row: f32 -> f16 convert + squared norm
__global__ void vq_prep(const float* __restrict__ cb,
                        _Float16* __restrict__ cbh,
                        float* __restrict__ wnorm) {
    __shared__ float red[128];
    int row = blockIdx.x;
    int tid = threadIdx.x;
    const float* src = cb + (size_t)row * DDIM;
    _Float16*    dst = cbh + (size_t)row * DDIM;
    float s = 0.0f;
#pragma unroll
    for (int j = 0; j < 4; ++j) {
        int e = tid + j * 128;
        float v = src[e];
        dst[e] = (_Float16)v;
        s += v * v;
    }
    red[tid] = s;
    __syncthreads();
    for (int off = 64; off > 0; off >>= 1) {
        if (tid < off) red[tid] += red[tid + off];
        __syncthreads();
    }
    if (tid == 0) wnorm[row] = red[0];
}

// ---------------------------------------------------------------------------
// main: 64 rows per block (wave w owns rows w*16..w*16+15, A in registers),
// B tiles double-buffered in LDS via async global->LDS DMA, shared by 4 waves.
__global__ void __launch_bounds__(128)
vq_main(const float* __restrict__ ze,
        const float* __restrict__ cb,
        const _Float16* __restrict__ cbh,
        const float* __restrict__ wnorm,
        float* __restrict__ out_zq,      // [N*D]
        float* __restrict__ out_k,       // [N] (float-coded indices)
        int*   __restrict__ used,
        float* __restrict__ loss_accum) {
    __shared__ _Float16 btile[2][TILEH];   // 2 x 16 KB B tiles
    __shared__ int      kidx[64];
    __shared__ float    lred[128];

    const int tid   = threadIdx.x;
    const int lane  = tid & 31;
    const int wv    = tid >> 5;       // wave 0..3
    const int lmod  = lane & 15;      // 0..15
    const int lhi   = lane >> 4;      // 0/1
    const size_t row0  = (size_t)blockIdx.x * 64;        // block's first row
    const size_t rowW  = row0 + (size_t)wv * 16;         // this wave's rows

    // ---- load this lane's A fragments: f32 global -> f16 registers ----
    // lane holds row (rowW + lmod); per k-step two 8-half chunks at
    // kk*32 + lhi*8 and kk*32 + lhi*8 + 16 (ISA 16-bit A layout).
    v16h afrag[DDIM / 32];
    {
        const float* xrow = ze + (rowW + lmod) * DDIM + lhi * 8;
#pragma unroll
        for (int kk = 0; kk < DDIM / 32; ++kk) {
            const float* p = xrow + kk * 32;
            f4 f0 = *(const f4*)(p);
            f4 f1 = *(const f4*)(p + 4);
            f4 g0 = *(const f4*)(p + 16);
            f4 g1 = *(const f4*)(p + 20);
            h4 c0 = __builtin_convertvector(f0, h4);
            h4 c1 = __builtin_convertvector(f1, h4);
            h4 c2 = __builtin_convertvector(g0, h4);
            h4 c3 = __builtin_convertvector(g1, h4);
            half8 lo = __builtin_shufflevector(c0, c1, 0,1,2,3,4,5,6,7);
            half8 hi = __builtin_shufflevector(c2, c3, 0,1,2,3,4,5,6,7);
            afrag[kk] = __builtin_shufflevector(lo, hi, 0,1,2,3,4,5,6,7,
                                                        8,9,10,11,12,13,14,15);
        }
    }

    // ---- B tile staging (16 KB = 1024 x 16B chunks, 8 per thread) ----
    auto stage = [&](int buf, int t) {
        char* gsrc = (char*)(cbh + (size_t)t * TILEH);   // const cast for builtin
        char* ldst = (char*)&btile[buf][0];
#if HAVE_ASYNC_LDS
#pragma unroll
        for (int j = 0; j < 8; ++j) {
            int c = tid + j * 128;
            __builtin_amdgcn_global_load_async_to_lds_b128(
                (gas1_v4i)(gsrc + c * 16),
                (las3_v4i)(ldst + c * 16),
                0, 0);
        }
#else
#pragma unroll
        for (int j = 0; j < 8; ++j) {
            int c = tid + j * 128;
            *(f4*)(ldst + c * 16) = *(const f4*)(gsrc + c * 16);
        }
#endif
    };
    auto wait_async = []() {
#if HAVE_ASYNC_LDS
        __builtin_amdgcn_s_wait_asynccnt(0);
#endif
    };

    // ---- WMMA scan over all code tiles (double-buffered B) ----
    float bestv[8];
    int   besti[8];
#pragma unroll
    for (int i = 0; i < 8; ++i) { bestv[i] = 3.0e38f; besti[i] = 0; }

    stage(0, 0);
    for (int t = 0; t < NTILES; ++t) {
        const int cur = t & 1;
        wait_async();        // my async chunks of buffer `cur` have landed
        __syncthreads();     // everyone's chunks landed; prev compute done
        if (t + 1 < NTILES) stage(cur ^ 1, t + 1);

        const int code = t * 16 + lmod;
        const _Float16* bbase = &btile[cur][lmod * DDIM + lhi * 8];
        v8f acc = {0.f, 0.f, 0.f, 0.f, 0.f, 0.f, 0.f, 0.f};
#pragma unroll
        for (int kk = 0; kk < DDIM / 32; ++kk) {
            half8 blo = *(const half8*)(bbase + kk * 32);
            half8 bhi = *(const half8*)(bbase + kk * 32 + 16);
            v16h b = __builtin_shufflevector(blo, bhi, 0,1,2,3,4,5,6,7,
                                                       8,9,10,11,12,13,14,15);
            acc = __builtin_amdgcn_wmma_f32_16x16x32_f16(
                      false, afrag[kk], false, b, (short)0, acc, false, false);
        }
        const float wn = wnorm[code];
#pragma unroll
        for (int i = 0; i < 8; ++i) {
            float d = wn - 2.0f * acc[i];   // ||w||^2 - 2 x.w  (||x||^2 const)
            if (d < bestv[i]) { bestv[i] = d; besti[i] = code; }
        }
    }

    // ---- argmin reduce across the 16 lanes of each half-wave ----
#pragma unroll
    for (int i = 0; i < 8; ++i) {
        float v = bestv[i];
        int   ix = besti[i];
#pragma unroll
        for (int m = 1; m < 16; m <<= 1) {
            float ov = __shfl_xor(v, m, 32);
            int   oi = __shfl_xor(ix, m, 32);
            if (ov < v || (ov == v && oi < ix)) { v = ov; ix = oi; }
        }
        bestv[i] = v; besti[i] = ix;
    }
    if (lmod == 0) {
        // lane 0 holds rows 0..7 of this wave's tile, lane 16 rows 8..15
#pragma unroll
        for (int i = 0; i < 8; ++i) {
            const int r  = lhi * 8 + i;          // row within wave tile
            const int ix = besti[i];
            kidx[wv * 16 + r] = ix;
            out_k[rowW + r]   = (float)ix;
            used[ix] = 1;
        }
    }
    __syncthreads();

    // ---- gather z_q (== z_q_st numerically) + commitment-loss partial ----
    float ls = 0.0f;
    for (int e = tid; e < 64 * DDIM; e += 128) {
        const int r = e >> 9;          // /512
        const int d = e & (DDIM - 1);
        const int k = kidx[r];
        const float zq = cb[(size_t)k * DDIM + d];
        const float x  = ze[(row0 + r) * DDIM + d];
        out_zq[(row0 + r) * DDIM + d] = zq;
        const float df = x - zq;
        ls += df * df;
    }
    lred[tid] = ls;
    __syncthreads();
    for (int off = 64; off > 0; off >>= 1) {
        if (tid < off) lred[tid] += lred[tid + off];
        __syncthreads();
    }
    if (tid == 0) atomicAdd(loss_accum, lred[0]);
}

// ---------------------------------------------------------------------------
__global__ void vq_finalize(const int* __restrict__ used,
                            const float* __restrict__ loss_accum,
                            float* __restrict__ out_tail) {  // [loss, util]
    __shared__ int red[256];
    int tid = threadIdx.x;
    int s = 0;
    for (int i = tid; i < KCODES; i += 256) s += used[i];
    red[tid] = s;
    __syncthreads();
    for (int off = 128; off > 0; off >>= 1) {
        if (tid < off) red[tid] += red[tid + off];
        __syncthreads();
    }
    if (tid == 0) {
        out_tail[0] = BETA * (*loss_accum) / (float)ND;   // vq_loss
        out_tail[1] = (float)red[0] / (float)KCODES;      // utilization
    }
}

// ---------------------------------------------------------------------------
extern "C" void kernel_launch(void* const* d_in, const int* in_sizes, int n_in,
                              void* d_out, int out_size, void* d_ws, size_t ws_size,
                              hipStream_t stream) {
    const float* ze = (const float*)d_in[0];
    const float* cb = (const float*)d_in[1];
    float* out = (float*)d_out;

    char* ws = (char*)d_ws;
    float*    ws_loss  = (float*)(ws + WS_LOSS);
    int*      ws_used  = (int*)(ws + WS_USED);
    float*    ws_wnorm = (float*)(ws + WS_WNORM);
    _Float16* ws_cbh   = (_Float16*)(ws + WS_CBH);

    float* out_zq   = out;               // [N*D]
    float* out_k    = out + ND;          // [N]
    float* out_tail = out + ND + NROWS;  // [2]

    vq_init<<<(KCODES + 255) / 256, 256, 0, stream>>>(ws_loss, ws_used);
    vq_prep<<<KCODES, 128, 0, stream>>>(cb, ws_cbh, ws_wnorm);
    vq_main<<<NROWS / 64, 128, 0, stream>>>(ze, cb, ws_cbh, ws_wnorm,
                                            out_zq, out_k, ws_used, ws_loss);
    vq_finalize<<<1, 256, 0, stream>>>(ws_used, ws_loss, out_tail);
}
